// binary_conv_71382356459826
// MI455X (gfx1250) — compile-verified, matrix operand order checked by reading, not run on hardware
//
#include <hip/hip_runtime.h>
#include <hip/hip_bf16.h>

typedef __attribute__((ext_vector_type(16))) _Float16 v16h;
typedef __attribute__((ext_vector_type(8)))  _Float16 v8h;
typedef __attribute__((ext_vector_type(8)))  float    v8f;

#define CH_IN   128
#define NF      256
#define WIN     1152
#define NSEL    16
#define KTOT    (WIN * NSEL)       // 18432: K of the flattened GEMM
#define MT      32                 // output rows per workgroup
#define NT      128                // output cols per workgroup
#define PCHUNK  4                  // window elements per K-chunk
#define KCHUNK  (PCHUNK * NSEL)    // 64 K per chunk = 2 WMMA K-steps
#define NCHUNK  (WIN / PCHUNK)     // 288 chunks
#define LSTRIDE 72                 // LDS row stride in halves (64 + 8 pad)

// Gather one window element of the SAME-padded 3x3 patch (channel-major p = c*9+kh*3+kw).
__device__ __forceinline__ float load_patch(const float* __restrict__ x,
                                            int n, int h, int w, int p) {
  int c  = p / 9;
  int r  = p - 9 * c;
  int kh = r / 3;
  int kw = r - 3 * kh;
  int hh = h + kh - 1;
  int ww = w + kw - 1;
  if ((unsigned)hh >= 32u || (unsigned)ww >= 32u) return 0.0f;
  return x[(size_t)(((n * 32 + hh) * 32 + ww)) * CH_IN + c];
}

// Build WsumT[f][p*16+sel] = g*(sign(wf[sel,p,f]) + sign(wf[sel+16,p,f])) in f16.
__global__ __launch_bounds__(256) void prep_wsum(const float* __restrict__ w,
                                                 const float* __restrict__ gamma,
                                                 _Float16* __restrict__ wsumT) {
  int idx = blockIdx.x * 256 + threadIdx.x;   // f-major, kk fastest -> coalesced stores
  int f   = idx / KTOT;
  int kk  = idx - f * KTOT;
  int p   = kk >> 4;
  int sel = kk & 15;
  int c   = p / 9;
  int r   = p - 9 * c;
  int kh  = r / 3;
  int kw  = r - 3 * kh;
  float g = fabsf(gamma[0]);
  size_t off1 = ((size_t)(((sel * 3 + kh) * 3 + kw) * CH_IN + c)) * NF + f;
  size_t off2 = off1 + (size_t)16 * 3 * 3 * CH_IN * NF;
  float v1 = w[off1], v2 = w[off2];
  float s1 = (v1 > 0.f) ? 1.f : ((v1 < 0.f) ? -1.f : 0.f);
  float s2 = (v2 > 0.f) ? 1.f : ((v2 < 0.f) ? -1.f : 0.f);
  wsumT[idx] = (_Float16)(g * (s1 + s2));
}

// Fused LUT-selector build + WMMA GEMM with async-to-LDS double-buffered B tiles.
// Grid: 256 WGs = 128 M-tiles(32 rows) x 2 N-tiles(128 cols); 256 threads = 8 waves.
// Wave split: wave_m = wave&1 (16-row subtile), wave_n = wave>>1 (32-col subtile).
__global__ __launch_bounds__(256) void lut_wmma(const float* __restrict__ x,
                                                const int* __restrict__ rmap,
                                                const _Float16* __restrict__ wsumT,
                                                float* __restrict__ out) {
  __shared__ __align__(16) _Float16 sA[MT * LSTRIDE];          // 4.6 KB
  __shared__ __align__(16) _Float16 sB[2][NT * LSTRIDE];       // 2 x 18 KB

  const int tid    = threadIdx.x;
  const int lane   = tid & 31;
  const int wave   = tid >> 5;
  const int wave_m = wave & 1;
  const int wave_n = wave >> 1;            // 0..3
  const int mtile  = (int)blockIdx.x >> 1;
  const int ntile  = (int)blockIdx.x & 1;
  const int m_base = mtile * MT;
  const int n_base = ntile * NT;

  const int lrow  = lane & 15;
  const int khalf = (lane & 16) ? 8 : 0;   // 16-bit WMMA operand lane split

  // LDS byte offsets (low 32 bits of the flat shared address == LDS offset)
  const unsigned sB0 = (unsigned)(uintptr_t)(&sB[0][0]);
  const unsigned sB1 = (unsigned)(uintptr_t)(&sB[1][0]);
  const unsigned bufOff[2] = {sB0, sB1};

  v8f acc[2];
#pragma unroll
  for (int i = 0; i < 2; ++i) {
    v8f z = {0.f, 0.f, 0.f, 0.f, 0.f, 0.f, 0.f, 0.f};
    acc[i] = z;
  }

  // ---- async B-copy: 128 cols x 64 halves = 1024 x 16B granules, 4 per thread ----
  auto issue_b_copy = [&](int kkbase, unsigned ldsBase) {
#pragma unroll
    for (int j = 0; j < 4; ++j) {
      int g   = tid + j * 256;             // 0..1023
      int col = g >> 3;                    // 0..127
      int seg = g & 7;                     // 0..7 (8 halves each)
      const _Float16* gp = wsumT + (size_t)(n_base + col) * KTOT + kkbase + seg * 8;
      unsigned lp = ldsBase + (unsigned)((col * LSTRIDE + seg * 8) * 2);
      unsigned long long ga = (unsigned long long)(uintptr_t)gp;
      asm volatile("global_load_async_to_lds_b128 %0, %1, off"
                   :: "v"(lp), "v"(ga)
                   : "memory");
    }
  };

  // prologue: chunk 0 -> buffer 0
  issue_b_copy(0, bufOff[0]);

  for (int ci = 0; ci < NCHUNK; ++ci) {
    const int buf = ci & 1;
    __syncthreads();   // sA and sB[buf^1] fully consumed by previous iteration

    // ---- build A tile: 32 rows x (4 p x 16 sel) f16 coefficients ----
    if (tid < MT * PCHUNK) {               // 128 (row,p) pairs
      int row = tid & (MT - 1);
      int pp  = tid >> 5;
      int p   = ci * PCHUNK + pp;
      int m   = m_base + row;
      int n   = m >> 10;
      int h   = (m >> 5) & 31;
      int ww  = m & 31;
      float P  = load_patch(x, n, h, ww, p);
      float q0 = load_patch(x, n, h, ww, rmap[p]);
      float q1 = load_patch(x, n, h, ww, rmap[WIN + p]);
      float q2 = load_patch(x, n, h, ww, rmap[2 * WIN + p]);
      float xa[2];
      xa[0] = P > 0.f ? P : 0.f;           // x_pos
      xa[1] = P < 0.f ? -P : 0.f;          // x_neg
      float m0[2], m1[2], m2[2];
      m0[0] = q0 > 0.f ? 1.f : (q0 < 0.f ? 0.f : 0.5f); m0[1] = 1.f - m0[0];
      m1[0] = q1 > 0.f ? 1.f : (q1 < 0.f ? 0.f : 0.5f); m1[1] = 1.f - m1[0];
      m2[0] = q2 > 0.f ? 1.f : (q2 < 0.f ? 0.f : 0.5f); m2[1] = 1.f - m2[0];
      _Float16* dst = &sA[row * LSTRIDE + pp * 16];
#pragma unroll
      for (int sel = 0; sel < 16; ++sel) {
        float cf = xa[sel >> 3] * m0[(sel >> 2) & 1] * m1[(sel >> 1) & 1] * m2[sel & 1];
        dst[sel] = (_Float16)cf;
      }
    }

    // ---- overlap: kick off next chunk's B copy, then wait for this chunk's ----
    if (ci + 1 < NCHUNK) {
      issue_b_copy((ci + 1) * KCHUNK, bufOff[buf ^ 1]);
      asm volatile("s_wait_asynccnt 0x4" ::: "memory");  // oldest 4 (chunk ci) retired
    } else {
      asm volatile("s_wait_asynccnt 0x0" ::: "memory");
    }
    __syncthreads();   // A tile visible; every wave's chunk-ci B copies landed

    // ---- WMMA: 2 K-steps x 2 N-fragments per wave ----
#pragma unroll
    for (int ks = 0; ks < KCHUNK; ks += 32) {
      const _Float16* ap = &sA[(wave_m * 16 + lrow) * LSTRIDE + ks + khalf];
      v8h alo = *reinterpret_cast<const v8h*>(ap);
      v8h ahi = *reinterpret_cast<const v8h*>(ap + 16);
      v16h afrag = __builtin_shufflevector(alo, ahi,
          0, 1, 2, 3, 4, 5, 6, 7, 8, 9, 10, 11, 12, 13, 14, 15);
#pragma unroll
      for (int nf = 0; nf < 2; ++nf) {
        int col = wave_n * 32 + nf * 16 + lrow;          // local col in N-tile
        const _Float16* bp = &sB[buf][col * LSTRIDE + ks + khalf];
        v8h blo = *reinterpret_cast<const v8h*>(bp);
        v8h bhi = *reinterpret_cast<const v8h*>(bp + 16);
        v16h bfrag = __builtin_shufflevector(blo, bhi,
            0, 1, 2, 3, 4, 5, 6, 7, 8, 9, 10, 11, 12, 13, 14, 15);
        acc[nf] = __builtin_amdgcn_wmma_f32_16x16x32_f16(
            /*neg_a=*/false, afrag, /*neg_b=*/false, bfrag,
            /*c_mod=*/(short)0, acc[nf], /*reuse_a=*/false, /*reuse_b=*/false);
      }
    }
  }

  // ---- store C/D: lane<16 -> rows 0..7 (v0..7), lane>=16 -> rows 8..15 ----
  const int mrow0 = m_base + wave_m * 16 + ((lane >> 4) << 3);
#pragma unroll
  for (int nf = 0; nf < 2; ++nf) {
    int col = n_base + wave_n * 32 + nf * 16 + lrow;
#pragma unroll
    for (int v = 0; v < 8; ++v) {
      out[(size_t)(mrow0 + v) * NF + col] = acc[nf][v];
    }
  }
}

extern "C" void kernel_launch(void* const* d_in, const int* in_sizes, int n_in,
                              void* d_out, int out_size, void* d_ws, size_t ws_size,
                              hipStream_t stream) {
  const float* x     = (const float*)d_in[0];   // [4,32,32,128]
  const float* w     = (const float*)d_in[1];   // [32,3,3,128,256]
  const float* gamma = (const float*)d_in[2];   // [1]
  const int*   rmap  = (const int*)d_in[3];     // [3,1152]
  float*       out   = (float*)d_out;           // [4,32,32,256]
  _Float16*    wsumT = (_Float16*)d_ws;         // [256][18432] f16 = 9.4 MB

  // 1) effective binarized weights, level-merged and transposed for B-fragment loads
  prep_wsum<<<KTOT * NF / 256, 256, 0, stream>>>(w, gamma, wsumT);
  // 2) fused selector-build + WMMA GEMM (M=4096, K=18432, N=256)
  lut_wmma<<<(4096 / MT) * (NF / NT), 256, 0, stream>>>(x, rmap, wsumT, out);
}
